// multi_head_attention_64561948393657
// MI455X (gfx1250) — matrix operand order repaired and checked by reading, not compiled, not run on hardware
//
#include <hip/hip_runtime.h>

// ---------------- problem constants ----------------
#define BATCH 1024
#define LSEQ  30      // title length
#define DDIM  300     // model dim
#define NHEAD 20
#define HDIM  20
#define LP    32      // padded L (2 tiles of 16)
#define DP    320     // padded D (10 k-steps of 32)
#define KSTEPS 10     // DP / 32
#define NTD   19      // n-tiles of 16 covering 304 >= 300
#define NB    2       // batches fused per workgroup

#define NQFRAG (NHEAD * KSTEPS * NTD)   // 3800 pre-swizzled Q fragments
#define NVFRAG (NHEAD * KSTEPS * 2)     // 400 pre-swizzled V fragments
#define FRAGEL 512                      // bf16 elements per fragment (32 lanes x 16)

typedef __attribute__((ext_vector_type(16))) __bf16 v16bf;
typedef __attribute__((ext_vector_type(8)))  __bf16 v8bf;
typedef __attribute__((ext_vector_type(8)))  float  v8f;

#define WMMA_BF16(a, b, c) \
  __builtin_amdgcn_wmma_f32_16x16x32_bf16(false, (a), false, (b), (short)0, (c), false, false)

__device__ __forceinline__ v8f zero8() {
  v8f z = {0.f, 0.f, 0.f, 0.f, 0.f, 0.f, 0.f, 0.f};
  return z;
}

// A fragment (16x32 bf16) from LDS row-major [m][k] (contiguous k order):
// lane holds row m0+(L%16); elems 0..15 -> K=k0+16*half+e.
__device__ __forceinline__ v16bf load_A_lds(const __bf16* base, int stride,
                                            int m0, int k0, int lane) {
  int m = m0 + (lane & 15);
  int hf = lane >> 4;
  const __bf16* p = base + m * stride + k0 + 16 * hf;
  v8bf lo = *(const v8bf*)(p);
  v8bf hi = *(const v8bf*)(p + 8);
  return __builtin_shufflevector(lo, hi, 0, 1, 2, 3, 4, 5, 6, 7,
                                 8, 9, 10, 11, 12, 13, 14, 15);
}

// B fragment where LDS holds B^T row-major: elem (k,n) at base[n*stride+k].
__device__ __forceinline__ v16bf load_Bt_lds(const __bf16* base, int stride,
                                             int k0, int n0, int lane) {
  int n = n0 + (lane & 15);
  int hf = lane >> 4;
  const __bf16* p = base + n * stride + k0 + 16 * hf;
  v8bf lo = *(const v8bf*)(p);
  v8bf hi = *(const v8bf*)(p + 8);
  return __builtin_shufflevector(lo, hi, 0, 1, 2, 3, 4, 5, 6, 7,
                                 8, 9, 10, 11, 12, 13, 14, 15);
}

// B fragment from LDS row-major [k][n] (strided per-element gather).
__device__ __forceinline__ v16bf load_Brm_lds(const __bf16* base, int stride,
                                              int k0, int n0, int lane) {
  int n = n0 + (lane & 15);
  int kb = k0 + 16 * (lane >> 4);
  v16bf b;
#pragma unroll
  for (int e = 0; e < 16; ++e) b[e] = base[(kb + e) * stride + n];
  return b;
}

// B fragment from pre-swizzled global bf16 fragments: 32 contiguous bytes/lane.
__device__ __forceinline__ v16bf load_Bsw(const __bf16* frags, int frag, int lane) {
  const __bf16* p = frags + (size_t)frag * FRAGEL + lane * 16;
  v8bf lo = *(const v8bf*)(p);
  v8bf hi = *(const v8bf*)(p + 8);
  return __builtin_shufflevector(lo, hi, 0, 1, 2, 3, 4, 5, 6, 7,
                                 8, 9, 10, 11, 12, 13, 14, 15);
}

// B fragment from global f32 row-major [k][ld], bounds-checked (fallback path).
__device__ __forceinline__ v16bf load_Bg_f32(const float* base, int ld, int kmax,
                                             int nmax, int k0, int n0, int lane) {
  int n = n0 + (lane & 15);
  int kb = k0 + 16 * (lane >> 4);
  v16bf b;
#pragma unroll
  for (int e = 0; e < 16; ++e) {
    int k = kb + e;
    float v = (k < kmax && n < nmax) ? base[(size_t)k * ld + n] : 0.0f;
    b[e] = (__bf16)v;
  }
  return b;
}

// C/D (16x16 f32): VGPR v -> M = m0 + v + 8*half, N = n0 + (lane%16).
__device__ __forceinline__ void store_C_bf16(__bf16* base, int stride, int m0,
                                             int n0, v8f c, int lane) {
  int n = n0 + (lane & 15);
  int hf = lane >> 4;
#pragma unroll
  for (int v = 0; v < 8; ++v) base[(m0 + v + 8 * hf) * stride + n] = (__bf16)c[v];
}

__device__ __forceinline__ void store_C_f32(float* base, int stride, int m0,
                                            int n0, v8f c, int lane) {
  int n = n0 + (lane & 15);
  int hf = lane >> 4;
#pragma unroll
  for (int v = 0; v < 8; ++v) base[(m0 + v + 8 * hf) * stride + n] = c[v];
}

// ---------------- TDM (Tensor Data Mover) 1D global->LDS copy ----------------
#if __has_builtin(__builtin_amdgcn_tensor_load_to_lds)
#define HAVE_TDM 1
#else
#define HAVE_TDM 0
#endif

#if HAVE_TDM
typedef unsigned int u32x4 __attribute__((ext_vector_type(4)));
typedef int i32x4 __attribute__((ext_vector_type(4)));
typedef int i32x8 __attribute__((ext_vector_type(8)));

// Copy ndwords dwords from gsrc (global) to lds_off (LDS byte offset) as a
// 1D tensor tile. D# fields per CDNA5 ISA.
__device__ __forceinline__ void tdm_copy_to_lds(unsigned lds_off,
                                                const void* gsrc,
                                                unsigned ndwords) {
  unsigned long long ga = (unsigned long long)gsrc;
  u32x4 g0;
  g0.x = 1u;                                              // count=1 valid user D#
  g0.y = lds_off;                                         // lds_addr
  g0.z = (unsigned)(ga & 0xFFFFFFFFu);                    // global_addr[31:0]
  g0.w = (unsigned)((ga >> 32) & 0x01FFFFFFu) | 0x80000000u; // addr[56:32]|type=2
  i32x8 g1;
  g1[0] = (int)(2u << 16);                                // data_size = 4 bytes
  g1[1] = (int)((ndwords & 0xFFFFu) << 16);               // tensor_dim0 lo16
  g1[2] = (int)(((ndwords >> 16) & 0xFFFFu) | (1u << 16)); // dim0 hi16 | dim1=1
  g1[3] = (int)((ndwords & 0xFFFFu) << 16);               // dim1 hi=0 | tile_dim0
  g1[4] = 0;                                              // tile_dim1/2 unused
  g1[5] = (int)ndwords;                                   // dim0_stride lo32
  g1[6] = 0;
  g1[7] = 0;
  i32x4 g2 = {0, 0, 0, 0};
  i32x4 g3 = {0, 0, 0, 0};
#if defined(__clang_major__) && __clang_major__ >= 23
  i32x8 g4 = {0, 0, 0, 0, 0, 0, 0, 0};
  __builtin_amdgcn_tensor_load_to_lds(g0, g1, g2, g3, g4, 0);
#else
  __builtin_amdgcn_tensor_load_to_lds(g0, g1, g2, g3, 0);
#endif
}

__device__ __forceinline__ void tdm_wait0() {
#if __has_builtin(__builtin_amdgcn_s_wait_tensorcnt)
  __builtin_amdgcn_s_wait_tensorcnt(0);
#else
  asm volatile("s_wait_tensorcnt 0x0" ::: "memory");
#endif
}
#endif  // HAVE_TDM

// ---------------- prep kernel: convert + swizzle into workspace ----------------
// blocks [0,10240): Y f32 -> Ybf bf16 [1024][32][320], zero padded (4 elem/thread)
// blocks [10240,10715): Q -> Qs fragments (one wave per fragment)
// blocks [10715,10765): V -> Vs fragments
#define PREP_YBLK 10240
#define PREP_QBLK ((NQFRAG + 7) / 8)
#define PREP_VBLK ((NVFRAG + 7) / 8)

__global__ __launch_bounds__(256, 1)
void prep_convert(const float* __restrict__ y, const float* __restrict__ Q,
                  const float* __restrict__ V, __bf16* __restrict__ Ybf,
                  __bf16* __restrict__ Qs, __bf16* __restrict__ Vs) {
  const int bid = blockIdx.x;
  const int tid = threadIdx.x;
  if (bid < PREP_YBLK) {
    size_t base = (size_t)bid * 1024 + (size_t)tid * 4;
#pragma unroll
    for (int i = 0; i < 4; ++i) {
      size_t p = base + i;                 // < 1024*32*320 == 10,485,760
      int b = (int)(p / (LP * DP));
      int rem = (int)(p % (LP * DP));
      int r = rem / DP, d = rem % DP;
      float v = (r < LSEQ && d < DDIM)
                    ? y[((size_t)b * LSEQ + r) * DDIM + d] : 0.0f;
      Ybf[p] = (__bf16)v;
    }
  } else if (bid < PREP_YBLK + PREP_QBLK) {
    int fb = (bid - PREP_YBLK) * 8 + (tid >> 5);
    int lane = tid & 31;
    if (fb < NQFRAG) {
      int nt = fb % NTD;
      int ks = (fb / NTD) % KSTEPS;
      int h = fb / (NTD * KSTEPS);
      int n = nt * 16 + (lane & 15);
      int kb = ks * 32 + 16 * (lane >> 4);
      __bf16* dst = Qs + (size_t)fb * FRAGEL + lane * 16;
#pragma unroll
      for (int e = 0; e < 16; ++e) {
        int k = kb + e;
        float v = (k < DDIM && n < DDIM)
                      ? Q[((size_t)h * DDIM + k) * DDIM + n] : 0.0f;
        dst[e] = (__bf16)v;
      }
    }
  } else {
    int fb = (bid - PREP_YBLK - PREP_QBLK) * 8 + (tid >> 5);
    int lane = tid & 31;
    if (fb < NVFRAG) {
      int nt = fb & 1;
      int ks = (fb >> 1) % KSTEPS;
      int h = (fb >> 1) / KSTEPS;
      int n = nt * 16 + (lane & 15);
      int kb = ks * 32 + 16 * (lane >> 4);
      __bf16* dst = Vs + (size_t)fb * FRAGEL + lane * 16;
#pragma unroll
      for (int e = 0; e < 16; ++e) {
        int k = kb + e;
        float v = (k < DDIM && n < HDIM)
                      ? V[((size_t)h * DDIM + k) * HDIM + n] : 0.0f;
        dst[e] = (__bf16)v;
      }
    }
  }
}

// ---------------- shared memory ----------------
struct __align__(16) Smem {
  __bf16 Y[NB][LP][DP];    // staged Y_b (bf16)                    40960 B
  __bf16 A1[NB][LP][DP];   // a1, later reused for H0              40960 B
  float  S[NB][LP][LP];    // raw scores                            8192 B
  __bf16 Al[NB][LP][LP];   // alpha (bf16 A operand)                4096 B
};                         // 94208 B (< 320 KB WGP LDS)

// ---------------- main fused attention kernel (workspace path) ----------------
__global__ __launch_bounds__(256, 1)
void attn_fused_wmma(const __bf16* __restrict__ Ybf, const __bf16* __restrict__ Qs,
                     const __bf16* __restrict__ Vs, const int* __restrict__ seq,
                     float* __restrict__ out) {
  __shared__ Smem sm;
  const int tid  = threadIdx.x;
  const int wave = tid >> 5;
  const int lane = tid & 31;
  const int h    = blockIdx.x % NHEAD;
  const int b0   = (blockIdx.x / NHEAD) * NB;

  // ---- Phase 0: stage Y_b tiles into LDS (TDM async-tensor DMA if available),
  //      zero the 304..319 pad columns of A1 ----
#if HAVE_TDM
  if (wave == 0) {
#pragma unroll
    for (int b = 0; b < NB; ++b)
      tdm_copy_to_lds((unsigned)(size_t)(void*)&sm.Y[b][0][0],
                      Ybf + (size_t)(b0 + b) * LP * DP, (LP * DP * 2) / 4);
  }
  tdm_wait0();
#else
  {
    const uint4* src = (const uint4*)(Ybf + (size_t)b0 * LP * DP);
    uint4* dst = (uint4*)&sm.Y[0][0][0];
    for (int i = tid; i < (NB * LP * DP) / 8; i += 256) dst[i] = src[i];
  }
#endif
  for (int i = tid; i < NB * LP * 16; i += 256) {
    int b = i / (LP * 16);
    int r = (i >> 4) % LP;
    sm.A1[b][r][304 + (i & 15)] = (__bf16)0.0f;
  }
  __syncthreads();

  // ---- GEMM1: A1[b] = Y[b](30x300) x Q_h(300x300), B from swizzled frags ----
  for (int nt = wave; nt < NTD; nt += 8) {
    int n0 = nt * 16;
    v8f acc[NB][2];
#pragma unroll
    for (int b = 0; b < NB; ++b)
#pragma unroll
      for (int mt = 0; mt < 2; ++mt) acc[b][mt] = zero8();
    for (int ks = 0; ks < KSTEPS; ++ks) {
      int frag = (h * KSTEPS + ks) * NTD + nt;
      v16bf bf = load_Bsw(Qs, frag, lane);
      if (ks + 1 < KSTEPS)
        __builtin_prefetch(Qs + (size_t)(frag + NTD) * FRAGEL + lane * 16, 0, 1);
#pragma unroll
      for (int b = 0; b < NB; ++b)
#pragma unroll
        for (int mt = 0; mt < 2; ++mt) {
          v16bf af = load_A_lds(&sm.Y[b][0][0], DP, mt * 16, ks * 32, lane);
          acc[b][mt] = WMMA_BF16(af, bf, acc[b][mt]);
        }
    }
#pragma unroll
    for (int b = 0; b < NB; ++b)
#pragma unroll
      for (int mt = 0; mt < 2; ++mt)
        store_C_bf16(&sm.A1[b][0][0], DP, mt * 16, n0, acc[b][mt], lane);
  }
  __syncthreads();

  // ---- GEMM2: S[b] = A1[b](30x300) x Y[b]^T(300x30), 8 tile-jobs, 1/wave ----
  {
    int b = wave >> 2, mt = (wave >> 1) & 1, nt = wave & 1;
    v8f acc = zero8();
    for (int ks = 0; ks < KSTEPS; ++ks) {
      v16bf af = load_A_lds(&sm.A1[b][0][0], DP, mt * 16, ks * 32, lane);
      v16bf bf = load_Bt_lds(&sm.Y[b][0][0], DP, ks * 32, nt * 16, lane);
      acc = WMMA_BF16(af, bf, acc);
    }
    store_C_f32(&sm.S[b][0][0], LP, mt * 16, nt * 16, acc, lane);
  }
  __syncthreads();

  // ---- exp + masked-denominator normalization (bug-for-bug faithful) ----
  if (tid < NB * LP) {
    int b = tid >> 5, l = tid & 31;
    const int* sq = seq + (size_t)(b0 + b) * LSEQ;
    float denom = 1e-6f;
    float ex[LP];
#pragma unroll
    for (int m = 0; m < LP; ++m) {
      float e = 0.0f;
      if (m < LSEQ) {
        e = __expf(sm.S[b][l][m]);     // no max subtraction, per reference
        if (sq[m] > 0) denom += e;     // denominator masked...
      }
      ex[m] = e;                       // ...numerator NOT masked
    }
    float inv = __builtin_amdgcn_rcpf(denom);
#pragma unroll
    for (int m = 0; m < LP; ++m) sm.Al[b][l][m] = (__bf16)(ex[m] * inv);
  }
  __syncthreads();

  // ---- GEMM3: H0[b] = Al[b](30x30) x Y[b](30x300), overwrite A1 ----
  for (int j = wave; j < NB * 2 * NTD; j += 8) {
    int b = j / (2 * NTD);
    int r = j % (2 * NTD);
    int mt = r / NTD, nt = r % NTD;
    v16bf af = load_A_lds(&sm.Al[b][0][0], LP, mt * 16, 0, lane);
    v16bf bf = load_Brm_lds(&sm.Y[b][0][0], DP, 0, nt * 16, lane);
    v8f acc = WMMA_BF16(af, bf, zero8());
    store_C_bf16(&sm.A1[b][0][0], DP, mt * 16, nt * 16, acc, lane);
  }
  __syncthreads();

  // ---- GEMM4: out = H0(30x300) x V_h(300x20), store directly ----
  {
    int b = wave >> 2, mt = (wave >> 1) & 1, nt = wave & 1;
    int n0 = nt * 16;
    v8f acc = zero8();
    for (int ks = 0; ks < KSTEPS; ++ks) {
      v16bf af = load_A_lds(&sm.A1[b][0][0], DP, mt * 16, ks * 32, lane);
      v16bf bf = load_Bsw(Vs, (h * KSTEPS + ks) * 2 + nt, lane);
      acc = WMMA_BF16(af, bf, acc);
    }
    int n = n0 + (lane & 15);
    int hf = lane >> 4;
    if (n < HDIM) {
#pragma unroll
      for (int v = 0; v < 8; ++v) {
        int l = mt * 16 + v + 8 * hf;
        if (l < LSEQ)
          out[((size_t)(b0 + b) * LSEQ + l) * (NHEAD * HDIM) + h * HDIM + n] = acc[v];
      }
    }
  }
}

// ---------------- fallback kernel (no workspace): round-1 version ----------------
__global__ __launch_bounds__(256, 1)
void attn_fused_wmma_nows(const float* __restrict__ y, const int* __restrict__ seq,
                          const float* __restrict__ Q, const float* __restrict__ V,
                          float* __restrict__ out) {
  __shared__ Smem sm;
  const int tid  = threadIdx.x;
  const int wave = tid >> 5;
  const int lane = tid & 31;
  const int h    = blockIdx.x % NHEAD;
  const int b0   = (blockIdx.x / NHEAD) * NB;
  const float* Qh = Q + (size_t)h * DDIM * DDIM;
  const float* Vh = V + (size_t)h * DDIM * HDIM;

  for (int idx = tid; idx < NB * LP * DP; idx += 256) {
    int b = idx / (LP * DP);
    int rem = idx % (LP * DP);
    int r = rem / DP, d = rem % DP;
    float v = 0.0f;
    if (r < LSEQ && d < DDIM) v = y[((size_t)(b0 + b) * LSEQ + r) * DDIM + d];
    sm.Y[b][r][d]  = (__bf16)v;
    sm.A1[b][r][d] = (__bf16)0.0f;
  }
  __syncthreads();

  for (int nt = wave; nt < NTD; nt += 8) {
    int n0 = nt * 16;
    v8f acc[NB][2];
#pragma unroll
    for (int b = 0; b < NB; ++b)
#pragma unroll
      for (int mt = 0; mt < 2; ++mt) acc[b][mt] = zero8();
    for (int ks = 0; ks < KSTEPS; ++ks) {
      int k0 = ks * 32;
      v16bf bf = load_Bg_f32(Qh, DDIM, DDIM, DDIM, k0, n0, lane);
#pragma unroll
      for (int b = 0; b < NB; ++b)
#pragma unroll
        for (int mt = 0; mt < 2; ++mt) {
          v16bf af = load_A_lds(&sm.Y[b][0][0], DP, mt * 16, k0, lane);
          acc[b][mt] = WMMA_BF16(af, bf, acc[b][mt]);
        }
    }
#pragma unroll
    for (int b = 0; b < NB; ++b)
#pragma unroll
      for (int mt = 0; mt < 2; ++mt)
        store_C_bf16(&sm.A1[b][0][0], DP, mt * 16, n0, acc[b][mt], lane);
  }
  __syncthreads();

  {
    int b = wave >> 2, mt = (wave >> 1) & 1, nt = wave & 1;
    v8f acc = zero8();
    for (int ks = 0; ks < KSTEPS; ++ks) {
      v16bf af = load_A_lds(&sm.A1[b][0][0], DP, mt * 16, ks * 32, lane);
      v16bf bf = load_Bt_lds(&sm.Y[b][0][0], DP, ks * 32, nt * 16, lane);
      acc = WMMA_BF16(af, bf, acc);
    }
    store_C_f32(&sm.S[b][0][0], LP, mt * 16, nt * 16, acc, lane);
  }
  __syncthreads();

  if (tid < NB * LP) {
    int b = tid >> 5, l = tid & 31;
    const int* sq = seq + (size_t)(b0 + b) * LSEQ;
    float denom = 1e-6f;
    float ex[LP];
#pragma unroll
    for (int m = 0; m < LP; ++m) {
      float e = 0.0f;
      if (m < LSEQ) {
        e = __expf(sm.S[b][l][m]);
        if (sq[m] > 0) denom += e;
      }
      ex[m] = e;
    }
    float inv = __builtin_amdgcn_rcpf(denom);
#pragma unroll
    for (int m = 0; m < LP; ++m) sm.Al[b][l][m] = (__bf16)(ex[m] * inv);
  }
  __syncthreads();

  for (int j = wave; j < NB * 2 * NTD; j += 8) {
    int b = j / (2 * NTD);
    int r = j % (2 * NTD);
    int mt = r / NTD, nt = r % NTD;
    v16bf af = load_A_lds(&sm.Al[b][0][0], LP, mt * 16, 0, lane);
    v16bf bf = load_Brm_lds(&sm.Y[b][0][0], DP, 0, nt * 16, lane);
    v8f acc = WMMA_BF16(af, bf, zero8());
    store_C_bf16(&sm.A1[b][0][0], DP, mt * 16, nt * 16, acc, lane);
  }
  __syncthreads();

  {
    int b = wave >> 2, mt = (wave >> 1) & 1, nt = wave & 1;
    int n0 = nt * 16;
    v8f acc = zero8();
    for (int ks = 0; ks < KSTEPS; ++ks) {
      v16bf af = load_A_lds(&sm.A1[b][0][0], DP, mt * 16, ks * 32, lane);
      v16bf bf = load_Bg_f32(Vh, HDIM, DDIM, HDIM, ks * 32, n0, lane);
      acc = WMMA_BF16(af, bf, acc);
    }
    int n = n0 + (lane & 15);
    int hf = lane >> 4;
    if (n < HDIM) {
#pragma unroll
      for (int v = 0; v < 8; ++v) {
        int l = mt * 16 + v + 8 * hf;
        if (l < LSEQ)
          out[((size_t)(b0 + b) * LSEQ + l) * (NHEAD * HDIM) + h * HDIM + n] = acc[v];
      }
    }
  }
}

extern "C" void kernel_launch(void* const* d_in, const int* in_sizes, int n_in,
                              void* d_out, int out_size, void* d_ws, size_t ws_size,
                              hipStream_t stream) {
  const float* y   = (const float*)d_in[0];
  const int*   seq = (const int*)  d_in[1];
  const float* Q   = (const float*)d_in[2];
  const float* V   = (const float*)d_in[3];
  float*       out = (float*)d_out;
  (void)in_sizes; (void)n_in; (void)out_size;

  const size_t needY = (size_t)BATCH * LP * DP * sizeof(__bf16);      // 20.97 MB
  const size_t needQ = (size_t)NQFRAG * FRAGEL * sizeof(__bf16);      //  3.89 MB
  const size_t needV = (size_t)NVFRAG * FRAGEL * sizeof(__bf16);      //  0.41 MB

  if (ws_size >= needY + needQ + needV) {
    __bf16* Ybf = (__bf16*)d_ws;
    __bf16* Qs  = (__bf16*)((char*)d_ws + needY);
    __bf16* Vs  = (__bf16*)((char*)d_ws + needY + needQ);
    prep_convert<<<PREP_YBLK + PREP_QBLK + PREP_VBLK, 256, 0, stream>>>(
        y, Q, V, Ybf, Qs, Vs);
    attn_fused_wmma<<<NHEAD * (BATCH / NB), 256, 0, stream>>>(Ybf, Qs, Vs, seq, out);
  } else {
    attn_fused_wmma_nows<<<NHEAD * (BATCH / NB), 256, 0, stream>>>(y, seq, Q, V, out);
  }
}
